// LGMRecModel_28157805592959
// MI455X (gfx1250) — matrix operand order repaired and checked by reading, not compiled, not run on hardware
//
#include <hip/hip_runtime.h>
#include <stdint.h>

// ---------------- problem constants ----------------
#define U_NUM    50000
#define I_NUM    30000
#define N_NODES_C 80000
#define EMB      64
#define HYP      32
#define FV_DIM   4096
#define FT_DIM   384
#define NNZ_A    2000000LL
#define NNZ_R    1000000LL
#define TAU_C    0.2f
#define ALPHA_C  0.2f

typedef __bf16 bf16_t;
typedef __attribute__((ext_vector_type(16))) bf16_t v16bf;
typedef __attribute__((ext_vector_type(8)))  float  v8f;

// ---------------- wave helpers (wave32) ----------------
__device__ inline float wave_sum(float v) {
#pragma unroll
    for (int o = 16; o; o >>= 1) v += __shfl_xor(v, o, 32);
    return v;
}
__device__ inline float wave_max(float v) {
#pragma unroll
    for (int o = 16; o; o >>= 1) v = fmaxf(v, __shfl_xor(v, o, 32));
    return v;
}

// ---------------- elementwise utility kernels ----------------
__global__ void k_fill0(float* p, long long n) {
    long long t = (long long)blockIdx.x * blockDim.x + threadIdx.x;
    if (t < n) p[t] = 0.0f;
}
__global__ void k_copy(float* dst, const float* src, long long n) {
    long long t = (long long)blockIdx.x * blockDim.x + threadIdx.x;
    if (t < n) dst[t] = src[t];
}
__global__ void k_add(float* dst, const float* src, long long n) {
    long long t = (long long)blockIdx.x * blockDim.x + threadIdx.x;
    if (t < n) dst[t] += src[t];
}
__global__ void k_scale(float* p, float s, long long n) {
    long long t = (long long)blockIdx.x * blockDim.x + threadIdx.x;
    if (t < n) p[t] *= s;
}
// dst[row*ldd + c] = src[row*lds + soff + c]
__global__ void k_copy_strided(const float* src, int lds, int soff,
                               float* dst, int ldd, long long rows, int cols) {
    long long t = (long long)blockIdx.x * blockDim.x + threadIdx.x;
    if (t >= rows * cols) return;
    long long row = t / cols; int c = (int)(t % cols);
    dst[row * (long long)ldd + c] = src[row * (long long)lds + soff + c];
}
// p[row*cols + c] *= s[row]
__global__ void k_rowscale(float* p, const float* s, long long rows, int cols) {
    long long t = (long long)blockIdx.x * blockDim.x + threadIdx.x;
    if (t >= rows * cols) return;
    p[t] *= s[t / cols];
}

// ---- build bf16 B panel pre-swizzled into the WMMA B-fragment register layout ----
// Logical B is [Kdim x 96] = [trs | hyp].  For K-chunk kc (32 rows), n-tile nt (16
// cols), lane L (half=L>>4, r=L&15), element j (0..15):
//   value = B[(kc*32 + (j < 8 ? 8*half + j : 16 + 8*half + (j - 8))) * 96 + nt*16 + r]
// stored at Bsw[ ((kc*6 + nt)*32 + L)*16 + j ]  -> each lane's fragment is 32
// contiguous, 32B-aligned bytes (two global_load_b128 per fragment).
__global__ void k_build_bsw(const float* __restrict__ trs, const float* __restrict__ hyp,
                            bf16_t* __restrict__ out, int Kdim) {
    int idx = blockIdx.x * blockDim.x + threadIdx.x;
    if (idx >= Kdim * 96) return;
    int j    = idx & 15;
    int L    = (idx >> 4) & 31;
    int rem  = idx >> 9;          // = kc*6 + nt
    int nt   = rem % 6;
    int kc   = rem / 6;
    int half = L >> 4, r = L & 15;
    int k = kc * 32 + (j < 8 ? 8 * half + j : 16 + 8 * half + (j - 8));
    int n = nt * 16 + r;
    float v = (n < 64) ? trs[(long long)k * 64 + n] : hyp[(long long)k * 32 + (n - 64)];
    out[idx] = (bf16_t)v;
}

// ---------------- WMMA GEMM: C[Mx96] = A[MxK](f32) * Bsw(bf16, swizzled) ----------
// M % 16 == 0, K % 32 == 0.  Each wave computes a 32x96 strip (2 M-tiles x 6 N-tiles).
__device__ inline v16bf load_a_frag(const float* ap) {
    float4 f0 = *reinterpret_cast<const float4*>(ap);
    float4 f1 = *reinterpret_cast<const float4*>(ap + 4);
    float4 f2 = *reinterpret_cast<const float4*>(ap + 16);
    float4 f3 = *reinterpret_cast<const float4*>(ap + 20);
    v16bf a;
    a[0]=(bf16_t)f0.x;  a[1]=(bf16_t)f0.y;  a[2]=(bf16_t)f0.z;  a[3]=(bf16_t)f0.w;
    a[4]=(bf16_t)f1.x;  a[5]=(bf16_t)f1.y;  a[6]=(bf16_t)f1.z;  a[7]=(bf16_t)f1.w;
    a[8]=(bf16_t)f2.x;  a[9]=(bf16_t)f2.y;  a[10]=(bf16_t)f2.z; a[11]=(bf16_t)f2.w;
    a[12]=(bf16_t)f3.x; a[13]=(bf16_t)f3.y; a[14]=(bf16_t)f3.z; a[15]=(bf16_t)f3.w;
    return a;
}

__global__ __launch_bounds__(256)
void gemm_wmma_bf16_n96(const float* __restrict__ A, const bf16_t* __restrict__ Bsw,
                        float* __restrict__ C, int M, int K) {
    const int tid   = threadIdx.x;
    const int lane  = tid & 31;
    const int half  = lane >> 4;
    const int r     = lane & 15;
    const int Mtiles = M >> 4;
    const int pair  = blockIdx.x * 8 + (tid >> 5);
    const int t0 = pair * 2, t1 = t0 + 1;
    if (t0 >= Mtiles) return;               // wave-uniform exit
    const bool have1 = (t1 < Mtiles);       // wave-uniform

    const float* Arow0 = A + (size_t)(t0 * 16 + r) * K + 8 * half;
    const float* Arow1 = have1 ? (A + (size_t)(t1 * 16 + r) * K + 8 * half) : Arow0;

    v8f acc0[6], acc1[6];
#pragma unroll
    for (int t = 0; t < 6; ++t) {
        v8f z = {0.f, 0.f, 0.f, 0.f, 0.f, 0.f, 0.f, 0.f};
        acc0[t] = z; acc1[t] = z;
    }

    const bf16_t* bb = Bsw + (size_t)lane * 16;
    for (int k0 = 0; k0 < K; k0 += 32) {
        const size_t kc = (size_t)(k0 >> 5);
        v16bf b[6];
#pragma unroll
        for (int nt = 0; nt < 6; ++nt)
            b[nt] = *reinterpret_cast<const v16bf*>(bb + (kc * 6 + nt) * 512);

        __builtin_prefetch(Arow0 + k0 + 32, 0, 0);       // global_prefetch next A slice
        v16bf a0 = load_a_frag(Arow0 + k0);
#pragma unroll
        for (int nt = 0; nt < 6; ++nt)
            acc0[nt] = __builtin_amdgcn_wmma_f32_16x16x32_bf16(
                false, a0, false, b[nt], (short)0, acc0[nt], false, false);

        if (have1) {                                      // wave-uniform
            __builtin_prefetch(Arow1 + k0 + 32, 0, 0);
            v16bf a1 = load_a_frag(Arow1 + k0);
#pragma unroll
            for (int nt = 0; nt < 6; ++nt)
                acc1[nt] = __builtin_amdgcn_wmma_f32_16x16x32_bf16(
                    false, a1, false, b[nt], (short)0, acc1[nt], false, false);
        }
    }

#pragma unroll
    for (int nt = 0; nt < 6; ++nt)
#pragma unroll
        for (int i = 0; i < 8; ++i)
            C[(size_t)(t0 * 16 + 8 * half + i) * 96 + nt * 16 + r] = acc0[nt][i];
    if (have1) {
#pragma unroll
        for (int nt = 0; nt < 6; ++nt)
#pragma unroll
            for (int i = 0; i < 8; ++i)
                C[(size_t)(t1 * 16 + 8 * half + i) * 96 + nt * 16 + r] = acc1[nt][i];
    }
}

// ---------------- SpMM via L2-resident f32 atomics ----------------
template <int C>
__global__ void spmm_atomic(const int* __restrict__ rows, const int* __restrict__ cols,
                            const float* __restrict__ vals, long long nnz,
                            const float* __restrict__ x, int ldx, int xoff,
                            float* __restrict__ out, int ldo) {
    long long t = (long long)blockIdx.x * blockDim.x + threadIdx.x;
    long long e = t / C;
    int c = (int)(t % C);
    if (e >= nnz) return;
    int rr = rows[e], cc = cols[e];
    float v = vals[e] * x[(size_t)cc * ldx + xoff + c];
    unsafeAtomicAdd(&out[(size_t)rr * ldo + c], v);   // global_atomic_add_f32 (no return)
}

// ---------------- gumbel softmax over rows of width 32 (one wave / row) -----------
__global__ void k_gumbel_softmax32(const float* __restrict__ logits, int ld, int off,
                                   float* __restrict__ out, int nrows, uint32_t seed) {
    int row = blockIdx.x * (blockDim.x >> 5) + (threadIdx.x >> 5);
    int lane = threadIdx.x & 31;
    if (row >= nrows) return;
    float x = logits[(size_t)row * ld + off + lane];
    uint32_t h = seed ^ (0x9E3779B1u * (uint32_t)row) ^ (0x85EBCA77u * (uint32_t)lane);
    h ^= h >> 16; h *= 0x7FEB352Du; h ^= h >> 15; h *= 0x846CA68Bu; h ^= h >> 16;
    float u = (float)(h >> 8) * (1.0f / 16777216.0f) + 5.96e-8f;
    float g = -logf(-logf(u));
    float y = (x + g) * (1.0f / TAU_C);
    float m = wave_max(y);
    float ex = expf(y - m);
    float s = wave_sum(ex);
    out[(size_t)row * 32 + lane] = ex / s;
}

// ---------------- dst[row] += alpha * normalize(src[row])  (K=64, wave/row) -------
__global__ void k_l2norm_addto(const float* __restrict__ src, float* __restrict__ dst,
                               int nrows, float alpha) {
    int row = blockIdx.x * (blockDim.x >> 5) + (threadIdx.x >> 5);
    int lane = threadIdx.x & 31;
    if (row >= nrows) return;
    float a = src[(size_t)row * 64 + lane];
    float b = src[(size_t)row * 64 + lane + 32];
    float ss = wave_sum(a * a + b * b);
    float inv = alpha / fmaxf(sqrtf(ss), 1e-12f);
    dst[(size_t)row * 64 + lane]      += a * inv;
    dst[(size_t)row * 64 + lane + 32] += b * inv;
}

// ---------------- lat[32x64] += w^T @ x  (w: N x 32, x: N x 64) ----------------
__global__ __launch_bounds__(256)
void k_att_lat(const float* __restrict__ w, const float* __restrict__ x, int ldx,
               float* __restrict__ lat, int nrows) {
    int n  = threadIdx.x & 63;
    int kb = (threadIdx.x >> 6) << 3;       // 0,8,16,24
    float acc[8] = {0.f, 0.f, 0.f, 0.f, 0.f, 0.f, 0.f, 0.f};
    for (int row = blockIdx.x; row < nrows; row += gridDim.x) {
        float xv = x[(size_t)row * ldx + n];
        const float* wr = w + (size_t)row * 32 + kb;
#pragma unroll
        for (int j = 0; j < 8; ++j) acc[j] += wr[j] * xv;
    }
#pragma unroll
    for (int j = 0; j < 8; ++j) unsafeAtomicAdd(&lat[(kb + j) * 64 + n], acc[j]);
}

// ---------------- out[row] = w[row,:]@lat ; ghe[row] += out[row]  ----------------
__global__ __launch_bounds__(256)
void k_apply_hyper(const float* __restrict__ w, const float* __restrict__ lat,
                   float* __restrict__ out, float* __restrict__ ghe, int nrows) {
    __shared__ float slat[32 * 64];
    int tid = threadIdx.x;
    for (int i = tid; i < 2048; i += 256) slat[i] = lat[i];
    __syncthreads();
    int lane = tid & 31;
    int row0 = blockIdx.x * 8 + (tid >> 5);
    for (int row = row0; row < nrows; row += gridDim.x * 8) {
        float wl = w[(size_t)row * 32 + lane];
        float a0 = 0.f, a1 = 0.f;
#pragma unroll
        for (int k = 0; k < 32; ++k) {
            float wk = __shfl(wl, k, 32);
            a0 += wk * slat[k * 64 + lane];
            a1 += wk * slat[k * 64 + lane + 32];
        }
        out[(size_t)row * 64 + lane]      = a0;
        out[(size_t)row * 64 + lane + 32] = a1;
        ghe[(size_t)row * 64 + lane]      += a0;
        ghe[(size_t)row * 64 + lane + 32] += a1;
    }
}

// ---------------- final: all = lge + ALPHA*l2norm(ghe); split users/items ---------
__global__ void k_final_combine(const float* __restrict__ lge, const float* __restrict__ ghe,
                                float* __restrict__ outU, float* __restrict__ outI,
                                int U, int Ntot) {
    int row = blockIdx.x * (blockDim.x >> 5) + (threadIdx.x >> 5);
    int lane = threadIdx.x & 31;
    if (row >= Ntot) return;
    float g0 = ghe[(size_t)row * 64 + lane];
    float g1 = ghe[(size_t)row * 64 + lane + 32];
    float ss = wave_sum(g0 * g0 + g1 * g1);
    float inv = ALPHA_C / fmaxf(sqrtf(ss), 1e-12f);
    float o0 = lge[(size_t)row * 64 + lane]      + g0 * inv;
    float o1 = lge[(size_t)row * 64 + lane + 32] + g1 * inv;
    float* dst = (row < U) ? (outU + (size_t)row * 64) : (outI + (size_t)(row - U) * 64);
    dst[lane] = o0; dst[lane + 32] = o1;
}

// ---------------- host orchestration ----------------
static inline char* bump(char*& p, size_t bytes) {
    char* r = p;
    p += (bytes + 255) & ~(size_t)255;
    return r;
}
static inline unsigned gblk(long long total, int bs) { return (unsigned)((total + bs - 1) / bs); }

extern "C" void kernel_launch(void* const* d_in, const int* in_sizes, int n_in,
                              void* d_out, int out_size, void* d_ws, size_t ws_size,
                              hipStream_t stream) {
    const float* Gu      = (const float*)d_in[0];
    const float* Gi      = (const float*)d_in[1];
    const float* feat_v  = (const float*)d_in[2];
    const float* feat_t  = (const float*)d_in[3];
    const float* trs_v   = (const float*)d_in[4];
    const float* trs_t   = (const float*)d_in[5];
    const float* hyp_v   = (const float*)d_in[6];
    const float* hyp_t   = (const float*)d_in[7];
    const float* inv_int = (const float*)d_in[8];
    const float* adj_vals= (const float*)d_in[9];
    const float* r_vals  = (const float*)d_in[10];
    const int*   adj_rows= (const int*)d_in[11];
    const int*   adj_cols= (const int*)d_in[12];
    const int*   r_rows  = (const int*)d_in[13];
    const int*   r_cols  = (const int*)d_in[14];

    const int U = U_NUM, I = I_NUM, N = N_NODES_C;
    const long long UO = (long long)U * 64, IO = (long long)I * 64, NO = (long long)N * 64;

    char* w = (char*)d_ws;
    bf16_t* Bv   = (bf16_t*)bump(w, (size_t)FV_DIM * 96 * sizeof(bf16_t));
    bf16_t* Bt   = (bf16_t*)bump(w, (size_t)FT_DIM * 96 * sizeof(bf16_t));
    float* Pv    = (float*)bump(w, (size_t)I * 96 * 4);   // [:, :64]=item_feats, [:,64:]=ih_logits
    float* Pt    = (float*)bump(w, (size_t)I * 96 * 4);
    float* uhl_v = (float*)bump(w, (size_t)U * 32 * 4);
    float* uhl_t = (float*)bump(w, (size_t)U * 32 * 4);
    float* ihg_v = (float*)bump(w, (size_t)I * 32 * 4);
    float* ihg_t = (float*)bump(w, (size_t)I * 32 * 4);
    float* uhg_v = (float*)bump(w, (size_t)U * 32 * 4);
    float* uhg_t = (float*)bump(w, (size_t)U * 32 * 4);
    float* bufA  = (float*)bump(w, (size_t)NO * 4);
    float* bufB  = (float*)bump(w, (size_t)NO * 4);
    float* cge   = (float*)bump(w, (size_t)NO * 4);
    float* lge   = (float*)bump(w, (size_t)NO * 4);
    float* mge   = (float*)bump(w, (size_t)NO * 4);
    float* mge2  = (float*)bump(w, (size_t)NO * 4);
    float* ghe   = (float*)bump(w, (size_t)NO * 4);
    float* lat   = (float*)bump(w, 32 * 64 * 4);

    // ---- 1) pre-swizzled bf16 B panels: [trs | hyp] in WMMA fragment layout ----
    k_build_bsw<<<gblk((long long)FV_DIM * 96, 256), 256, 0, stream>>>(trs_v, hyp_v, Bv, FV_DIM);
    k_build_bsw<<<gblk((long long)FT_DIM * 96, 256), 256, 0, stream>>>(trs_t, hyp_t, Bt, FT_DIM);

    // ---- 2) WMMA GEMMs: one HBM pass over each feature matrix ----
    const int Mtiles = I / 16;                 // 1875
    const int pairs  = (Mtiles + 1) / 2;       // 938 (32-row strips per wave)
    gemm_wmma_bf16_n96<<<gblk(pairs, 8), 256, 0, stream>>>(feat_v, Bv, Pv, I, FV_DIM);
    gemm_wmma_bf16_n96<<<gblk(pairs, 8), 256, 0, stream>>>(feat_t, Bt, Pt, I, FT_DIM);

    // ---- 3) uh_logits = R @ ih_logits ----
    k_fill0<<<gblk((long long)U * 32, 256), 256, 0, stream>>>(uhl_v, (long long)U * 32);
    k_fill0<<<gblk((long long)U * 32, 256), 256, 0, stream>>>(uhl_t, (long long)U * 32);
    spmm_atomic<32><<<gblk(NNZ_R * 32, 256), 256, 0, stream>>>(r_rows, r_cols, r_vals, NNZ_R, Pv, 96, 64, uhl_v, 32);
    spmm_atomic<32><<<gblk(NNZ_R * 32, 256), 256, 0, stream>>>(r_rows, r_cols, r_vals, NNZ_R, Pt, 96, 64, uhl_t, 32);

    // ---- 4) gumbel softmax (deterministic counter hash) ----
    k_gumbel_softmax32<<<gblk(I, 8), 256, 0, stream>>>(Pv, 96, 64, ihg_v, I, 0x42000001u);
    k_gumbel_softmax32<<<gblk(U, 8), 256, 0, stream>>>(uhl_v, 32, 0, uhg_v, U, 0x42000002u);
    k_gumbel_softmax32<<<gblk(I, 8), 256, 0, stream>>>(Pt, 96, 64, ihg_t, I, 0x42000101u);
    k_gumbel_softmax32<<<gblk(U, 8), 256, 0, stream>>>(uhl_t, 32, 0, uhg_t, U, 0x42000102u);

    // ---- 5) collaborative graph embeddings: cge = (ego + A ego + A^2 ego)/3 ----
    k_copy<<<gblk(UO, 256), 256, 0, stream>>>(bufA, Gu, UO);
    k_copy<<<gblk(IO, 256), 256, 0, stream>>>(bufA + UO, Gi, IO);
    k_copy<<<gblk(NO, 256), 256, 0, stream>>>(cge, bufA, NO);
    k_fill0<<<gblk(NO, 256), 256, 0, stream>>>(bufB, NO);
    spmm_atomic<64><<<gblk(NNZ_A * 64, 256), 256, 0, stream>>>(adj_rows, adj_cols, adj_vals, NNZ_A, bufA, 64, 0, bufB, 64);
    k_add<<<gblk(NO, 256), 256, 0, stream>>>(cge, bufB, NO);
    k_fill0<<<gblk(NO, 256), 256, 0, stream>>>(bufA, NO);
    spmm_atomic<64><<<gblk(NNZ_A * 64, 256), 256, 0, stream>>>(adj_rows, adj_cols, adj_vals, NNZ_A, bufB, 64, 0, bufA, 64);
    k_add<<<gblk(NO, 256), 256, 0, stream>>>(cge, bufA, NO);
    k_scale<<<gblk(NO, 256), 256, 0, stream>>>(cge, 1.0f / 3.0f, NO);
    k_copy<<<gblk(NO, 256), 256, 0, stream>>>(lge, cge, NO);

    // ---- 6) modality graph embeddings, lge += l2norm(A @ mge) per modality ----
    const float* Ps[2] = {Pv, Pt};
    for (int m = 0; m < 2; ++m) {
        k_fill0<<<gblk(UO, 256), 256, 0, stream>>>(mge, UO);
        spmm_atomic<64><<<gblk(NNZ_R * 64, 256), 256, 0, stream>>>(r_rows, r_cols, r_vals, NNZ_R, Ps[m], 96, 0, mge, 64);
        k_rowscale<<<gblk(UO, 256), 256, 0, stream>>>(mge, inv_int, U, 64);
        k_copy_strided<<<gblk(IO, 256), 256, 0, stream>>>(Ps[m], 96, 0, mge + UO, 64, I, 64);
        k_fill0<<<gblk(NO, 256), 256, 0, stream>>>(mge2, NO);
        spmm_atomic<64><<<gblk(NNZ_A * 64, 256), 256, 0, stream>>>(adj_rows, adj_cols, adj_vals, NNZ_A, mge, 64, 0, mge2, 64);
        k_l2norm_addto<<<gblk(N, 8), 256, 0, stream>>>(mge2, lge, N, 1.0f);
    }

    // ---- 7) hypergraph path ----
    k_fill0<<<gblk(NO, 256), 256, 0, stream>>>(ghe, NO);
    float* outs = (float*)d_out;
    float* out_u[2] = {outs + UO + IO,       outs + 2 * (UO + IO)};
    float* out_i[2] = {outs + 2 * UO + IO,   outs + 3 * UO + 2 * IO};
    const float* ihg[2] = {ihg_v, ihg_t};
    const float* uhg[2] = {uhg_v, uhg_t};
    for (int m = 0; m < 2; ++m) {
        k_fill0<<<gblk(2048, 256), 256, 0, stream>>>(lat, 2048);
        k_att_lat<<<512, 256, 0, stream>>>(ihg[m], cge + UO, 64, lat, I);
        k_apply_hyper<<<gblk(I, 8), 256, 0, stream>>>(ihg[m], lat, out_i[m], ghe + UO, I);
        k_apply_hyper<<<gblk(U, 8), 256, 0, stream>>>(uhg[m], lat, out_u[m], ghe, U);
    }

    // ---- 8) all_embs = lge + ALPHA * l2norm(ghe); split users / items ----
    k_final_combine<<<gblk(N, 8), 256, 0, stream>>>(lge, ghe, outs, outs + UO, U, N);
}
